// Reasoning2D_53042846105816
// MI455X (gfx1250) — compile-verified, matrix operand order checked by reading, not run on hardware
//
#include <hip/hip_runtime.h>
#include <hip/hip_bf16.h>

// ---------------------------------------------------------------------------
// Reasoning2D on MI455X (gfx1250):
//   prep:      build effective 3x3x32x32 conv kernel + bias from parameters,
//              packed as f16 WMMA B-fragments in workspace.
//   conv_wmma: implicit-GEMM 3x3 VALID conv via v_wmma_f32_16x16x32_f16,
//              input staged f32->f16 into LDS per workgroup tile.
//              Output stores are non-temporal: the 264 MB output is streamed
//              once, so NT keeps the 192 MB L2 free for the 3x-reused input
//              halo rows (CDNA5 ISA 7.3 temporal hints).
// ---------------------------------------------------------------------------

typedef __attribute__((ext_vector_type(16))) _Float16 v16h;
typedef __attribute__((ext_vector_type(8)))  _Float16 v8h;
typedef __attribute__((ext_vector_type(4)))  _Float16 v4h;
typedef __attribute__((ext_vector_type(8)))  float    v8f;

#define BATCH   32
#define HIN     256
#define WIN     256
#define CIN     32
#define NOUT    32
#define HOUT    254
#define WOUT    254
#define TILE_W  64          // output pixels along W per workgroup
#define LDS_W   66          // TILE_W + 2 halo
#define NFRAG_BYTES (2 * 9 * 32 * 16 * (int)sizeof(_Float16))  // B-fragments

// ---------------------------------------------------------------------------
// prep: one block of 288 threads (9 waves). Thread (j, o) with j = kernel
// position (kh*3+kw), o = output channel. Reproduces the reference math:
//   comp softmax over C, pixel softmax over (kh,kw), normalization, bias.
// Writes kernel values as f16 directly in WMMA B-fragment lane/slot layout:
//   fragment index [ntile(2)][j(9)][lane(32)][slot(16)]
//   lane L<16: N = L, slots 0..7 -> K(c)=0..7, slots 8..15 -> c=16..23
//   lane L+16: N = L, slots 0..7 -> c=8..15,   slots 8..15 -> c=24..31
// ---------------------------------------------------------------------------
__global__ __launch_bounds__(288) void reasoning_prep(
    const float* __restrict__ rp,     // (2,3,3,32,32)
    const float* __restrict__ pixp,   // (3,3,1,32)
    const float* __restrict__ compp,  // (1,1,32,1)
    _Float16* __restrict__ bfrag,     // [2][9][32][16] f16
    float* __restrict__ bias)         // [32]
{
    __shared__ float s_comp[CIN];
    __shared__ float s_bias[9 * NOUT];

    const int tid = threadIdx.x;

    // component softmax over C (32 values), computed redundantly by 32 threads
    if (tid < CIN) {
        float m = -3.0e38f;
        for (int c = 0; c < CIN; ++c) m = fmaxf(m, compp[c]);
        float s = 0.0f;
        for (int c = 0; c < CIN; ++c) s += __expf(compp[c] - m);
        s_comp[tid] = __expf(compp[tid] - m) / s;
    }
    __syncthreads();

    const int j  = tid >> 5;    // kernel position 0..8
    const int o  = tid & 31;    // output channel

    // pixel softmax over the 9 spatial positions, per channel o
    float pm = -3.0e38f;
    for (int jj = 0; jj < 9; ++jj) pm = fmaxf(pm, pixp[jj * NOUT + o]);
    float ps = 0.0f;
    for (int jj = 0; jj < 9; ++jj) ps += __expf(pixp[jj * NOUT + o] - pm);
    const float pp = __expf(pixp[j * NOUT + o] - pm) / ps;

    // reasoning slices for this (kh,kw): rp[s][j][c][o]
    const float* r0 = rp + (size_t)(0 * 9 + j) * CIN * NOUT;
    const float* r1 = rp + (size_t)(1 * 9 + j) * CIN * NOUT;

    // normalization = sum_c (pos + neg) + eps
    float norm = 1e-7f;
    for (int c = 0; c < CIN; ++c) {
        const float p0  = r0[c * NOUT + o];
        const float p1  = r1[c * NOUT + o];
        const float cm  = s_comp[c];
        norm += p0 * cm + (1.0f - p0) * p1 * cm;
    }
    const float inv = pp / norm;

    // kernel values -> B-fragment layout; accumulate bias part
    const int tile   = o >> 4;
    const int lanelo = o & 15;
    _Float16* frag = bfrag + (size_t)(tile * 9 + j) * 32 * 16;

    float bpart = 0.0f;
    for (int c = 0; c < CIN; ++c) {
        const float p0  = r0[c * NOUT + o];
        const float p1  = r1[c * NOUT + o];
        const float cm  = s_comp[c];
        const float pos = p0 * cm;
        const float neg = (1.0f - p0) * p1 * cm;
        bpart += neg * inv;
        const float kval = (pos - neg) * inv;

        const int blk  = c >> 3;                        // 0..3
        const int lane = lanelo + ((blk & 1) ? 16 : 0); // odd blocks -> hi lanes
        const int slot = (c & 7) + ((blk >= 2) ? 8 : 0);
        frag[lane * 16 + slot] = (_Float16)kval;
    }

    s_bias[j * NOUT + o] = bpart;
    __syncthreads();
    if (j == 0) {
        float s = 0.0f;
        for (int jj = 0; jj < 9; ++jj) s += s_bias[jj * NOUT + o];
        bias[o] = s;
    }
}

// ---------------------------------------------------------------------------
// conv_wmma: grid (ceil(254/64), 254, 32), block 128 (4 wave32).
// GEMM view: M = 64 output pixels (x), N = 32 channels, K = 9*32 = 288.
// Wave w: N-tile = w>>1, M-tile pair = (w&1)*2 + {0,1}.
// ---------------------------------------------------------------------------
__global__ __launch_bounds__(128) void conv_wmma(
    const float* __restrict__ in,      // (32,256,256,32) f32
    const _Float16* __restrict__ bfrag,
    const float* __restrict__ bias,
    float* __restrict__ out)           // (32,254,254,32) f32
{
    __shared__ __align__(16) _Float16 smem[3 * LDS_W * CIN];  // 12672 B

    const int b   = blockIdx.z;
    const int y0  = blockIdx.y;
    const int x0  = blockIdx.x * TILE_W;
    const int tid = threadIdx.x;

    // ---- stage 3 x 66 x 32 input halo tile into LDS as f16 ----
    // 3*66*8 = 1584 float4 quads, 128 threads strided
    for (int t = tid; t < 3 * LDS_W * 8; t += 128) {
        const int r   = t / (LDS_W * 8);
        const int rem = t - r * (LDS_W * 8);
        const int px  = rem >> 3;
        const int cq  = rem & 7;
        int gx = x0 + px; if (gx > WIN - 1) gx = WIN - 1;   // clamp tail halo
        const float4 v = *(const float4*)(
            in + ((((size_t)b * HIN + (y0 + r)) * WIN + gx) * CIN + cq * 4));
        v4h h;
        h.x = (_Float16)v.x; h.y = (_Float16)v.y;
        h.z = (_Float16)v.z; h.w = (_Float16)v.w;
        *(v4h*)(smem + (r * LDS_W + px) * CIN + cq * 4) = h;   // ds_store_b64
    }

    const int wave  = tid >> 5;
    const int lane  = tid & 31;
    const int ntile = wave >> 1;          // 0 -> channels 0..15, 1 -> 16..31
    const int mbase = (wave & 1) * 32;    // pixel offset of this wave's M pair
    const int lrow  = lane & 15;
    const int hi16  = (lane >> 4) & 1;

    // ---- load the 9 pre-packed B-fragments (2x global_load_b128 each) ----
    v16h bf[9];
    const _Float16* bp = bfrag + ((size_t)ntile * 9 * 32 + lane) * 16;
    #pragma unroll
    for (int j = 0; j < 9; ++j) {
        v8h lo = *(const v8h*)(bp + (size_t)j * 32 * 16);
        v8h hi = *(const v8h*)(bp + (size_t)j * 32 * 16 + 8);
        bf[j] = __builtin_shufflevector(lo, hi, 0, 1, 2, 3, 4, 5, 6, 7,
                                                8, 9, 10, 11, 12, 13, 14, 15);
    }
    const float bv = bias[ntile * 16 + lrow];

    __syncthreads();

    // ---- K-loop: 9 kernel positions x K=32 channels, 2 M tiles per wave ----
    v8f acc0 = {};
    v8f acc1 = {};
    #pragma unroll
    for (int j = 0; j < 9; ++j) {
        const int kh = j / 3;
        const int kw = j - kh * 3;
        // A fragment, 16-bit A 16x32 layout (ISA 7.12.2):
        //   lanes 0-15:  row M=lane,    c in {0..7, 16..23}
        //   lanes 16-31: row M=lane-16, c in {8..15, 24..31}
        {
            const int px = mbase + lrow + kw;
            const _Float16* ap = smem + (kh * LDS_W + px) * CIN + hi16 * 8;
            v8h lo = *(const v8h*)ap;            // ds_load_b128
            v8h hi = *(const v8h*)(ap + 16);     // ds_load_b128
            v16h a = __builtin_shufflevector(lo, hi, 0, 1, 2, 3, 4, 5, 6, 7,
                                                     8, 9, 10, 11, 12, 13, 14, 15);
            acc0 = __builtin_amdgcn_wmma_f32_16x16x32_f16(
                false, a, false, bf[j], (short)0, acc0, false, false);
        }
        {
            const int px = mbase + 16 + lrow + kw;
            const _Float16* ap = smem + (kh * LDS_W + px) * CIN + hi16 * 8;
            v8h lo = *(const v8h*)ap;
            v8h hi = *(const v8h*)(ap + 16);
            v16h a = __builtin_shufflevector(lo, hi, 0, 1, 2, 3, 4, 5, 6, 7,
                                                     8, 9, 10, 11, 12, 13, 14, 15);
            acc1 = __builtin_amdgcn_wmma_f32_16x16x32_f16(
                false, a, false, bf[j], (short)0, acc1, false, false);
        }
    }

    // ---- epilogue: add bias, non-temporal stores (guard W tail) ----
    // C/D layout: VGPR r -> M = r + 8*hi16, N = lane&15.
    // NT: output is write-once streamed; keep L2 for reused input rows.
    const int oc = ntile * 16 + lrow;
    const size_t rowbase = (((size_t)b * HOUT + y0) * WOUT) * NOUT;
    #pragma unroll
    for (int r = 0; r < 8; ++r) {
        const int m  = r + hi16 * 8;
        const int x1 = x0 + mbase + m;
        if (x1 < WOUT)
            __builtin_nontemporal_store(acc0[r] + bv,
                out + rowbase + (size_t)x1 * NOUT + oc);
        const int x2 = x0 + mbase + 16 + m;
        if (x2 < WOUT)
            __builtin_nontemporal_store(acc1[r] + bv,
                out + rowbase + (size_t)x2 * NOUT + oc);
    }
}

// ---------------------------------------------------------------------------
extern "C" void kernel_launch(void* const* d_in, const int* in_sizes, int n_in,
                              void* d_out, int out_size, void* d_ws, size_t ws_size,
                              hipStream_t stream) {
    (void)in_sizes; (void)n_in; (void)out_size; (void)ws_size;

    const float* in    = (const float*)d_in[0];  // (32,256,256,32)
    const float* rp    = (const float*)d_in[1];  // (2,3,3,32,32)
    const float* pixp  = (const float*)d_in[2];  // (3,3,1,32)
    const float* compp = (const float*)d_in[3];  // (1,1,32,1)
    float* out = (float*)d_out;                  // (32,254,254,32)

    _Float16* bfrag = (_Float16*)d_ws;                       // 18432 B
    float*    bias  = (float*)((char*)d_ws + NFRAG_BYTES);   // 128 B

    reasoning_prep<<<1, 288, 0, stream>>>(rp, pixp, compp, bfrag, bias);

    dim3 grid((WOUT + TILE_W - 1) / TILE_W, HOUT, BATCH);    // (4, 254, 32)
    conv_wmma<<<grid, 128, 0, stream>>>(in, bfrag, bias, out);
}
